// RAST_53455162966710
// MI455X (gfx1250) — compile-verified, matrix operand order checked by reading, not run on hardware
//
#include <hip/hip_runtime.h>
#include <hip/hip_bf16.h>

#define NNODES 50000
#define NEDGES 300000
#define DIM    256
#define HID    128
#define IDIM   64
#define BATCH  4096
#define SEQ    50

typedef __attribute__((ext_vector_type(16))) __bf16 v16bf;
typedef __attribute__((ext_vector_type(8)))  __bf16 v8bf;
typedef __attribute__((ext_vector_type(8)))  float  v8f;

__device__ __forceinline__ v16bf ld16(const __bf16* lo, const __bf16* hi) {
  v8bf a = *(const v8bf*)lo;
  v8bf b = *(const v8bf*)hi;
  return __builtin_shufflevector(a, b, 0,1,2,3,4,5,6,7,8,9,10,11,12,13,14,15);
}
__device__ __forceinline__ float sigf(float x){ return 1.0f/(1.0f+__expf(-x)); }

// ---------------- GCN (memory-bound, scalar path) ----------------
__global__ void deg_kernel(const int* __restrict__ head, float* __restrict__ deg) {
  int e = blockIdx.x*256 + threadIdx.x;
  if (e >= NEDGES) return;
  atomicAdd(&deg[head[e]], 1.0f);
}

__global__ void scatter_kernel(const float* __restrict__ cur, const float* __restrict__ relw,
                               const int* __restrict__ head, const int* __restrict__ tail,
                               const int* __restrict__ etype, float* __restrict__ nxt) {
  long long tid = (long long)blockIdx.x*256 + threadIdx.x;
  if (tid >= (long long)NEDGES*64) return;
  int e = (int)(tid >> 6);
  int d = ((int)tid & 63) * 4;
  int h = head[e], tl = tail[e], r = etype[e];
  float4 mv = *(const float4*)(cur  + (size_t)tl*DIM + d);
  float4 rv = *(const float4*)(relw + (size_t)r *DIM + d);
  float* dst = nxt + (size_t)h*DIM + d;
  atomicAdd(dst+0, mv.x*rv.x);
  atomicAdd(dst+1, mv.y*rv.y);
  atomicAdd(dst+2, mv.z*rv.z);
  atomicAdd(dst+3, mv.w*rv.w);
}

__global__ void finalize_kernel(float* __restrict__ nxt, const float* __restrict__ deg,
                                float* __restrict__ acc) {
  long long i = (long long)blockIdx.x*256 + threadIdx.x;
  if (i >= (long long)NNODES*DIM) return;
  float dg = deg[i >> 8];
  dg = dg < 1.0f ? 1.0f : dg;
  float v = nxt[i] / dg;
  nxt[i] = v;        // becomes "out" (input of next hop)
  acc[i] += v;
}

__global__ void gather_kernel(const float* __restrict__ acc, const int* __restrict__ nidx,
                              float* __restrict__ structv) {
  int i = blockIdx.x*256 + threadIdx.x;
  if (i >= BATCH*DIM) return;
  int b = i >> 8, d = i & 255;
  structv[i] = acc[(size_t)nidx[b]*DIM + d] * (1.0f/3.0f);   // /(HOPS+1)
}

// ---------------- conversions / weight packing ----------------
__global__ void cvt_kernel(const float* __restrict__ src, __bf16* __restrict__ dst, int n) {
  int i = blockIdx.x*256 + threadIdx.x;
  if (i < n) dst[i] = (__bf16)src[i];
}

__global__ void wcomb_kernel(const float* __restrict__ Wih, const float* __restrict__ Whh,
                             __bf16* __restrict__ Wc) {     // [4H, IDIM+HID]
  int i = blockIdx.x*256 + threadIdx.x;
  if (i >= 512*192) return;
  int r = i / 192, k = i % 192;
  float v = (k < IDIM) ? Wih[r*IDIM + k] : Whh[r*HID + (k - IDIM)];
  Wc[i] = (__bf16)v;
}

__global__ void w2col_kernel(const float* __restrict__ W2, const float* __restrict__ b2,
                             float* __restrict__ w2col) {   // w2col[256] + b2sum at [256]
  int k = threadIdx.x;
  float s = 0.0f;
  for (int n = 0; n < DIM; ++n) s += W2[n*DIM + k];
  w2col[k] = s;
  if (k == 0) {
    float t = 0.0f;
    for (int i = 0; i < DIM; ++i) t += b2[i];
    w2col[DIM] = t;
  }
}

// ---------------- fused LSTM step: g = [x_t|h] @ [Wih|Whh]^T + b ----------------
__global__ void __launch_bounds__(128)
lstm_step_kernel(const __bf16* __restrict__ ts_bf,  // [B,S,I]
                 const __bf16* __restrict__ Wc,     // [512,192]
                 const float*  __restrict__ bias,   // [512]
                 const __bf16* __restrict__ h_in,   // [B,128]
                 __bf16*       __restrict__ h_out,  // [B,128]
                 float*        __restrict__ cst,    // [B,128]
                 float*        __restrict__ temporal, // [B,256] running sum of h
                 int t, int dir_off)
{
  const int lane  = threadIdx.x & 31;
  const int wave  = threadIdx.x >> 5;
  const int m0    = blockIdx.x * 16;
  const int n0    = blockIdx.y * 64 + wave * 16;
  const int arow  = m0 + (lane & 15);
  const int khalf = (lane >> 4) * 8;

  const __bf16* xrow = ts_bf + ((size_t)arow * SEQ + t) * IDIM;
  const __bf16* hrow = h_in  + (size_t)arow * HID;
  const int nr = n0 + (lane & 15);
  const __bf16* wi = Wc + (size_t)(0*HID + nr) * 192;
  const __bf16* wf = Wc + (size_t)(1*HID + nr) * 192;
  const __bf16* wg = Wc + (size_t)(2*HID + nr) * 192;
  const __bf16* wo = Wc + (size_t)(3*HID + nr) * 192;

  v8f ai = {}, af = {}, ag = {}, ao = {};
#pragma unroll
  for (int kk = 0; kk < 192; kk += 32) {
    int k0 = kk + khalf, k1 = kk + 16 + khalf;
    const __bf16* p0 = (k0 < IDIM) ? (xrow + k0) : (hrow + (k0 - IDIM));
    const __bf16* p1 = (k1 < IDIM) ? (xrow + k1) : (hrow + (k1 - IDIM));
    v16bf a = ld16(p0, p1);
    v16bf b;
    b  = ld16(wi + k0, wi + k1);
    ai = __builtin_amdgcn_wmma_f32_16x16x32_bf16(false, a, false, b, (short)0, ai, false, false);
    b  = ld16(wf + k0, wf + k1);
    af = __builtin_amdgcn_wmma_f32_16x16x32_bf16(false, a, false, b, (short)0, af, false, false);
    b  = ld16(wg + k0, wg + k1);
    ag = __builtin_amdgcn_wmma_f32_16x16x32_bf16(false, a, false, b, (short)0, ag, false, false);
    b  = ld16(wo + k0, wo + k1);
    ao = __builtin_amdgcn_wmma_f32_16x16x32_bf16(false, a, false, b, (short)0, ao, false, false);
  }

  const int n     = n0 + (lane & 15);
  const int mbase = m0 + ((lane >> 4) << 3);
  const float bi = bias[0*HID + n], bff = bias[1*HID + n];
  const float bg = bias[2*HID + n], bo  = bias[3*HID + n];
#pragma unroll
  for (int r = 0; r < 8; ++r) {
    int m = mbase + r;
    size_t ci = (size_t)m * HID + n;
    float gi = sigf(ai[r] + bi);
    float gf = sigf(af[r] + bff);
    float gg = tanhf(ag[r] + bg);
    float go = sigf(ao[r] + bo);
    float cn = gf * cst[ci] + gi * gg;
    float h  = go * tanhf(cn);
    cst[ci]  = cn;
    h_out[ci] = (__bf16)h;
    temporal[(size_t)m * DIM + dir_off + n] += h;
  }
}

// ---------------- gate GEMM + merge (K=512) ----------------
__global__ void __launch_bounds__(128)
gate_merge_kernel(const __bf16* __restrict__ catbf,   // [B,512]
                  const __bf16* __restrict__ Wg,      // [256,512]
                  const float*  __restrict__ gate_b,  // [256]
                  const float*  __restrict__ structv, // [B,256]
                  const float*  __restrict__ temporal,// [B,256] sums
                  __bf16*       __restrict__ merged)  // [B,256]
{
  const int lane  = threadIdx.x & 31;
  const int wave  = threadIdx.x >> 5;
  const int m0    = blockIdx.x * 16;
  const int n0    = blockIdx.y * 64 + wave * 16;
  const int khalf = (lane >> 4) * 8;
  const __bf16* arow = catbf + (size_t)(m0 + (lane & 15)) * 512;
  const __bf16* wrow = Wg    + (size_t)(n0 + (lane & 15)) * 512;
  v8f acc = {};
#pragma unroll
  for (int kk = 0; kk < 512; kk += 32) {
    v16bf a = ld16(arow + kk + khalf, arow + kk + 16 + khalf);
    v16bf b = ld16(wrow + kk + khalf, wrow + kk + 16 + khalf);
    acc = __builtin_amdgcn_wmma_f32_16x16x32_bf16(false, a, false, b, (short)0, acc, false, false);
  }
  const int n     = n0 + (lane & 15);
  const int mbase = m0 + ((lane >> 4) << 3);
  const float invS = 1.0f / (float)SEQ;
  const float bn = gate_b[n];
#pragma unroll
  for (int r = 0; r < 8; ++r) {
    int m = mbase + r;
    size_t idx = (size_t)m * DIM + n;
    float g  = sigf(acc[r] + bn);
    float mv = g * structv[idx] + (1.0f - g) * temporal[idx] * invS;
    merged[idx] = (__bf16)mv;
  }
}

// ---------------- risk layer 1 GEMM + tanh (K=256) ----------------
__global__ void __launch_bounds__(128)
risk1_kernel(const __bf16* __restrict__ merged,  // [B,256]
             const __bf16* __restrict__ W1,      // [256,256]
             const float*  __restrict__ b1,      // [256]
             __bf16*       __restrict__ risk1)   // [B,256]
{
  const int lane  = threadIdx.x & 31;
  const int wave  = threadIdx.x >> 5;
  const int m0    = blockIdx.x * 16;
  const int n0    = blockIdx.y * 64 + wave * 16;
  const int khalf = (lane >> 4) * 8;
  const __bf16* arow = merged + (size_t)(m0 + (lane & 15)) * DIM;
  const __bf16* wrow = W1     + (size_t)(n0 + (lane & 15)) * DIM;
  v8f acc = {};
#pragma unroll
  for (int kk = 0; kk < DIM; kk += 32) {
    v16bf a = ld16(arow + kk + khalf, arow + kk + 16 + khalf);
    v16bf b = ld16(wrow + kk + khalf, wrow + kk + 16 + khalf);
    acc = __builtin_amdgcn_wmma_f32_16x16x32_bf16(false, a, false, b, (short)0, acc, false, false);
  }
  const int n     = n0 + (lane & 15);
  const int mbase = m0 + ((lane >> 4) << 3);
  const float bn = b1[n];
#pragma unroll
  for (int r = 0; r < 8; ++r) {
    int m = mbase + r;
    risk1[(size_t)m * DIM + n] = (__bf16)tanhf(acc[r] + bn);
  }
}

// ---------------- cat builder / final reduction ----------------
__global__ void build_cat_kernel(const float* __restrict__ structv,
                                 const float* __restrict__ temporal,
                                 __bf16* __restrict__ catbf) {
  int i = blockIdx.x*256 + threadIdx.x;
  if (i >= BATCH*DIM) return;
  int m = i >> 8, d = i & 255;
  const float invS = 1.0f / (float)SEQ;
  catbf[(size_t)m*512 + d]       = (__bf16)structv[i];
  catbf[(size_t)m*512 + DIM + d] = (__bf16)(temporal[i] * invS);
}

__global__ void final_kernel(const __bf16* __restrict__ risk1,
                             const float*  __restrict__ w2col,
                             float* __restrict__ out) {
  int b = blockIdx.x*256 + threadIdx.x;
  if (b >= BATCH) return;
  const __bf16* row = risk1 + (size_t)b * DIM;
  float s = w2col[DIM];   // sum(b2)
  for (int k = 0; k < DIM; ++k) s += (float)row[k] * w2col[k];
  out[b] = sigf(s);
}

extern "C" void kernel_launch(void* const* d_in, const int* in_sizes, int n_in,
                              void* d_out, int out_size, void* d_ws, size_t ws_size,
                              hipStream_t stream) {
  (void)in_sizes; (void)n_in; (void)out_size; (void)ws_size;
  const float* ts       = (const float*)d_in[0];
  const float* node_emb = (const float*)d_in[1];
  const float* rel_w    = (const float*)d_in[2];
  const float* gate_W   = (const float*)d_in[3];
  const float* gate_b   = (const float*)d_in[4];
  const float* risk_W1  = (const float*)d_in[5];
  const float* risk_b1  = (const float*)d_in[6];
  const float* risk_W2  = (const float*)d_in[7];
  const float* risk_b2  = (const float*)d_in[8];
  const float* Wih_f    = (const float*)d_in[9];
  const float* Whh_f    = (const float*)d_in[10];
  const float* b_f      = (const float*)d_in[11];
  const float* Wih_b    = (const float*)d_in[12];
  const float* Whh_b    = (const float*)d_in[13];
  const float* b_b      = (const float*)d_in[14];
  const int*   edge_idx = (const int*)d_in[15];
  const int*   edge_typ = (const int*)d_in[16];
  const int*   node_idx = (const int*)d_in[17];
  const int* head = edge_idx;
  const int* tail = edge_idx + NEDGES;
  float* out = (float*)d_out;

  char* w = (char*)d_ws;
  auto alloc = [&](size_t bytes)->char* {
    char* p = w; w += (bytes + 255) & ~(size_t)255; return p;
  };
  float*  acc      = (float*) alloc(sizeof(float)*(size_t)NNODES*DIM);
  float*  bufA     = (float*) alloc(sizeof(float)*(size_t)NNODES*DIM);
  float*  bufB     = (float*) alloc(sizeof(float)*(size_t)NNODES*DIM);
  float*  deg      = (float*) alloc(sizeof(float)*NNODES);
  __bf16* ts_bf    = (__bf16*)alloc(2ull*BATCH*SEQ*IDIM);
  __bf16* WcF      = (__bf16*)alloc(2ull*512*192);
  __bf16* WcB      = (__bf16*)alloc(2ull*512*192);
  __bf16* h0       = (__bf16*)alloc(2ull*BATCH*HID);
  __bf16* h1       = (__bf16*)alloc(2ull*BATCH*HID);
  float*  cbuf     = (float*) alloc(4ull*BATCH*HID);
  float*  temporal = (float*) alloc(4ull*BATCH*DIM);
  float*  structv  = (float*) alloc(4ull*BATCH*DIM);
  __bf16* catbf    = (__bf16*)alloc(2ull*BATCH*2*DIM);
  __bf16* Wgbf     = (__bf16*)alloc(2ull*DIM*2*DIM);
  __bf16* W1bf     = (__bf16*)alloc(2ull*DIM*DIM);
  __bf16* merged   = (__bf16*)alloc(2ull*BATCH*DIM);
  __bf16* risk1    = (__bf16*)alloc(2ull*BATCH*DIM);
  float*  w2col    = (float*) alloc(4ull*(DIM+1));

  // zero-init accumulators (workspace is poisoned, not zeroed)
  hipMemsetAsync(deg,      0, sizeof(float)*NNODES, stream);
  hipMemsetAsync(bufA,     0, sizeof(float)*(size_t)NNODES*DIM, stream);
  hipMemsetAsync(bufB,     0, sizeof(float)*(size_t)NNODES*DIM, stream);
  hipMemsetAsync(temporal, 0, 4ull*BATCH*DIM, stream);
  hipMemsetAsync(cbuf,     0, 4ull*BATCH*HID, stream);
  hipMemsetAsync(h0,       0, 2ull*BATCH*HID, stream);
  hipMemcpyAsync(acc, node_emb, sizeof(float)*(size_t)NNODES*DIM,
                 hipMemcpyDeviceToDevice, stream);

  // ---- GCN: 2 hops of relation-weighted scatter-mean ----
  deg_kernel<<<(NEDGES+255)/256, 256, 0, stream>>>(head, deg);
  const long long scatN = (long long)NEDGES*64;
  const long long ndN   = (long long)NNODES*DIM;
  scatter_kernel<<<(unsigned)((scatN+255)/256), 256, 0, stream>>>(node_emb, rel_w, head, tail, edge_typ, bufA);
  finalize_kernel<<<(unsigned)((ndN+255)/256), 256, 0, stream>>>(bufA, deg, acc);
  scatter_kernel<<<(unsigned)((scatN+255)/256), 256, 0, stream>>>(bufA, rel_w, head, tail, edge_typ, bufB);
  finalize_kernel<<<(unsigned)((ndN+255)/256), 256, 0, stream>>>(bufB, deg, acc);
  gather_kernel<<<(BATCH*DIM+255)/256, 256, 0, stream>>>(acc, node_idx, structv);

  // ---- bf16 conversions / packing ----
  const int tsN = BATCH*SEQ*IDIM;
  cvt_kernel<<<(tsN+255)/256, 256, 0, stream>>>(ts, ts_bf, tsN);
  wcomb_kernel<<<(512*192+255)/256, 256, 0, stream>>>(Wih_f, Whh_f, WcF);
  wcomb_kernel<<<(512*192+255)/256, 256, 0, stream>>>(Wih_b, Whh_b, WcB);
  cvt_kernel<<<(DIM*2*DIM+255)/256, 256, 0, stream>>>(gate_W, Wgbf, DIM*2*DIM);
  cvt_kernel<<<(DIM*DIM+255)/256, 256, 0, stream>>>(risk_W1, W1bf, DIM*DIM);
  w2col_kernel<<<1, 256, 0, stream>>>(risk_W2, risk_b2, w2col);

  // ---- BiLSTM: fused WMMA step kernel, ping-pong h buffers ----
  dim3 lgrid(BATCH/16, HID/64);           // (256, 2), 128 threads = 4 waves
  for (int t = 0; t < SEQ; ++t) {
    __bf16* hin  = (t & 1) ? h1 : h0;
    __bf16* hout = (t & 1) ? h0 : h1;
    lstm_step_kernel<<<lgrid, 128, 0, stream>>>(ts_bf, WcF, b_f, hin, hout, cbuf, temporal, t, 0);
  }
  hipMemsetAsync(cbuf, 0, 4ull*BATCH*HID, stream);
  hipMemsetAsync(h0,   0, 2ull*BATCH*HID, stream);
  for (int s = 0; s < SEQ; ++s) {
    int t = SEQ - 1 - s;
    __bf16* hin  = (s & 1) ? h1 : h0;
    __bf16* hout = (s & 1) ? h0 : h1;
    lstm_step_kernel<<<lgrid, 128, 0, stream>>>(ts_bf, WcB, b_b, hin, hout, cbuf, temporal, t, HID);
  }

  // ---- gate + merge + risk head ----
  build_cat_kernel<<<(BATCH*DIM+255)/256, 256, 0, stream>>>(structv, temporal, catbf);
  dim3 ggrid(BATCH/16, DIM/64);           // (256, 4)
  gate_merge_kernel<<<ggrid, 128, 0, stream>>>(catbf, Wgbf, gate_b, structv, temporal, merged);
  risk1_kernel<<<ggrid, 128, 0, stream>>>(merged, W1bf, risk_b1, risk1);
  final_kernel<<<(BATCH+255)/256, 256, 0, stream>>>(risk1, w2col, out);
}
